// Encoder_61959198212422
// MI455X (gfx1250) — compile-verified, matrix operand order checked by reading, not run on hardware
//
#include <hip/hip_runtime.h>
#include <hip/hip_bf16.h>
#include <math.h>

typedef __bf16 bf16;
typedef bf16 v16bf __attribute__((ext_vector_type(16)));
typedef bf16 v8bf  __attribute__((ext_vector_type(8)));
typedef float v8f  __attribute__((ext_vector_type(8)));

#define LAYERS 4
#define DMODEL 1024
#define NHEAD 16
#define DHEAD 64
#define DFFN 4096
#define BATCH 2
#define SEQ 2048
#define NTOK (BATCH * SEQ)
#define MiB ((size_t)1 << 20)

// ---------------------------------------------------------------------------
// CDNA5 async global->LDS copy (ASYNCcnt path). LDS offset = low 32 bits of
// the generic pointer (ISA 10.2).
// ---------------------------------------------------------------------------
__device__ __forceinline__ void async_load_16B(const void* g, void* lds_ptr) {
  unsigned off = (unsigned)(size_t)lds_ptr;
  unsigned long long ga = (unsigned long long)(size_t)g;
  asm volatile("global_load_async_to_lds_b128 %0, %1, off"
               :: "v"(off), "v"(ga)
               : "memory");
}
__device__ __forceinline__ void wait_async0() {
  asm volatile("s_wait_asynccnt 0x0" ::: "memory");
}

// ---------------------------------------------------------------------------
// Fragment loader: 16-bit A/B fragment for v_wmma_*_16x16x32_bf16.
// Per ISA 7.12.2: lane L holds row/col (L&15); element e maps to
// K = (e>>3)*16 + ((L>>4)*8) + (e&7)  -> two contiguous 8xbf16 (16B) chunks.
// ---------------------------------------------------------------------------
__device__ __forceinline__ v16bf load_frag(const bf16* p, int kbase) {
  v8bf lo = *(const v8bf*)(p + kbase);
  v8bf hi = *(const v8bf*)(p + 16 + kbase);
  v16bf r;
#pragma unroll
  for (int i = 0; i < 8; ++i) { r[i] = lo[i]; r[i + 8] = hi[i]; }
  return r;
}

// ---------------------------------------------------------------------------
// GEMM: C[M,N] = A[M,K] (bf16, row-major) * W[N,K]^T (bf16, torch [out,in])
//       + bias[N]; optional exact GELU; fp32 or bf16 output.
// Block = 128 threads (4 waves). B tile (64x32) double-buffered in LDS via
// async global->LDS loads overlapped with WMMA on the previous tile.
// Each wave computes a 32x64 strip (2 row-tiles x 4 col-tiles = 8 WMMAs/step).
// ---------------------------------------------------------------------------
template <bool GELU_ACT, bool OUT_BF16>
__global__ __launch_bounds__(128, 1) void wmma_gemm_kernel(
    const bf16* __restrict__ A, const bf16* __restrict__ W,
    const float* __restrict__ bias,
    float* __restrict__ Cf, bf16* __restrict__ Cb,
    int M, int N, int K) {
  __shared__ bf16 bls[2][64 * 32];

  const int lane  = threadIdx.x & 31;
  const int wave  = threadIdx.x >> 5;
  const int l15   = lane & 15;
  const int kbase = (lane >> 4) * 8;
  const int m0    = blockIdx.y * 128 + wave * 32;
  const int n0    = blockIdx.x * 64;

  v8f acc[2][4];
#pragma unroll
  for (int rt = 0; rt < 2; ++rt)
#pragma unroll
    for (int n = 0; n < 4; ++n)
#pragma unroll
      for (int r = 0; r < 8; ++r) acc[rt][n][r] = 0.0f;

  const bf16* arow0 = A + (size_t)(m0 + l15) * K;
  const bf16* arow1 = A + (size_t)(m0 + 16 + l15) * K;

  // this thread's two 16B chunks of the 64x32 B tile
  const int chunk0 = threadIdx.x;        // 0..127
  const int chunk1 = threadIdx.x + 128;  // 128..255
  const int rN0 = chunk0 >> 2, kc0 = (chunk0 & 3) * 8;
  const int rN1 = chunk1 >> 2, kc1 = (chunk1 & 3) * 8;
  const bf16* wp0 = W + (size_t)(n0 + rN0) * K + kc0;
  const bf16* wp1 = W + (size_t)(n0 + rN1) * K + kc1;

  // prologue: stage tile for kk=0 into buffer 0
  async_load_16B(wp0, &bls[0][rN0 * 32 + kc0]);
  async_load_16B(wp1, &bls[0][rN1 * 32 + kc1]);
  wait_async0();
  __syncthreads();

  int cur = 0;
  for (int kk = 0; kk < K; kk += 32) {
    // stage next tile into the other buffer (overlaps with WMMA below)
    if (kk + 32 < K) {
      async_load_16B(wp0 + kk + 32, &bls[cur ^ 1][rN0 * 32 + kc0]);
      async_load_16B(wp1 + kk + 32, &bls[cur ^ 1][rN1 * 32 + kc1]);
      __builtin_prefetch(arow0 + kk + 32, 0, 1);
      __builtin_prefetch(arow1 + kk + 32, 0, 1);
    }

    v16bf a0 = load_frag(arow0 + kk, kbase);
    v16bf a1 = load_frag(arow1 + kk, kbase);
#pragma unroll
    for (int n = 0; n < 4; ++n) {
      v16bf bfrag = load_frag(&bls[cur][(n * 16 + l15) * 32], kbase);
      acc[0][n] = __builtin_amdgcn_wmma_f32_16x16x32_bf16(
          false, a0, false, bfrag, (short)0, acc[0][n], false, false);
      acc[1][n] = __builtin_amdgcn_wmma_f32_16x16x32_bf16(
          false, a1, false, bfrag, (short)0, acc[1][n], false, false);
    }

    wait_async0();
    __syncthreads();
    cur ^= 1;
  }

  const int rowoff = (lane >> 4) * 8;
#pragma unroll
  for (int rt = 0; rt < 2; ++rt) {
#pragma unroll
    for (int n = 0; n < 4; ++n) {
      const int col = n0 + n * 16 + l15;
      const float bv = bias[col];
#pragma unroll
      for (int r = 0; r < 8; ++r) {
        const int row = m0 + rt * 16 + rowoff + r;
        float y = acc[rt][n][r] + bv;
        if (GELU_ACT) y = 0.5f * y * (1.0f + erff(y * 0.70710678118f));
        if (OUT_BF16) Cb[(size_t)row * N + col] = (bf16)y;
        else          Cf[(size_t)row * N + col] = y;
      }
    }
  }
}

// ---------------------------------------------------------------------------
// Flash attention: one wave per (b, h, 16-query tile). Online softmax in
// registers (offset-major ds_bpermute reductions); V tile fetched with async
// global->LDS loads hidden under the QK^T WMMAs + softmax; P staged via 1KB
// LDS tile for the C-layout -> A-layout transpose.
// ---------------------------------------------------------------------------
__global__ __launch_bounds__(32, 1) void flash_attn_kernel(
    const bf16* __restrict__ Q, const bf16* __restrict__ Kx,
    const bf16* __restrict__ V, const int* __restrict__ mask,
    bf16* __restrict__ ctx) {
  __shared__ bf16 pls[16 * 32];   // P tile (A-layout source)
  __shared__ bf16 vls[32 * 64];   // V tile (k-major)

  const int lane  = threadIdx.x;
  const int l15   = lane & 15;
  const int half  = lane >> 4;
  const int kbase = half * 8;
  const int qt = blockIdx.x, h = blockIdx.y, b = blockIdx.z;
  const int q0 = qt * 16;
  const int colQ = h * DHEAD;

  const bf16* qrow = Q + (size_t)(b * SEQ + q0 + l15) * DMODEL + colQ;
  v16bf qf0 = load_frag(qrow, kbase);        // K-dims 0..31 of DH
  v16bf qf1 = load_frag(qrow + 32, kbase);   // K-dims 32..63

  v8f acc[4];
  float m8[8], l8[8];
#pragma unroll
  for (int r = 0; r < 8; ++r) { m8[r] = -1e30f; l8[r] = 0.0f; }
#pragma unroll
  for (int n = 0; n < 4; ++n)
#pragma unroll
    for (int r = 0; r < 8; ++r) acc[n][r] = 0.0f;

  const float scale = 0.125f;  // 1/sqrt(64)

  for (int kb0 = 0; kb0 < SEQ; kb0 += 32) {
    // ---- async stage of V tile (32 rows x 64 cols); completes under QK^T ----
    const bf16* vrow = V + (size_t)(b * SEQ + kb0 + lane) * DMODEL + colQ;
#pragma unroll
    for (int c = 0; c < 8; ++c)
      async_load_16B(vrow + c * 8, &vls[lane * 64 + c * 8]);

    // ---- scores S = Q K^T for 16 queries x 32 keys (two 16x16 tiles) ----
    const bf16* krow0 = Kx + (size_t)(b * SEQ + kb0 + l15) * DMODEL + colQ;
    const bf16* krow1 = Kx + (size_t)(b * SEQ + kb0 + 16 + l15) * DMODEL + colQ;
    v8f s0, s1;
#pragma unroll
    for (int r = 0; r < 8; ++r) { s0[r] = 0.0f; s1[r] = 0.0f; }
    s0 = __builtin_amdgcn_wmma_f32_16x16x32_bf16(false, qf0, false, load_frag(krow0, kbase),      (short)0, s0, false, false);
    s0 = __builtin_amdgcn_wmma_f32_16x16x32_bf16(false, qf1, false, load_frag(krow0 + 32, kbase), (short)0, s0, false, false);
    s1 = __builtin_amdgcn_wmma_f32_16x16x32_bf16(false, qf0, false, load_frag(krow1, kbase),      (short)0, s1, false, false);
    s1 = __builtin_amdgcn_wmma_f32_16x16x32_bf16(false, qf1, false, load_frag(krow1 + 32, kbase), (short)0, s1, false, false);

    const bool keep0 = mask[b * SEQ + kb0 + l15] != 0;
    const bool keep1 = mask[b * SEQ + kb0 + 16 + l15] != 0;

    // ---- online softmax (row of C-frag elem r is r + 8*half) ----
    float a0[8], a1[8], mx[8], rs[8];
#pragma unroll
    for (int r = 0; r < 8; ++r) {
      a0[r] = keep0 ? s0[r] * scale : -1e30f;
      a1[r] = keep1 ? s1[r] * scale : -1e30f;
      mx[r] = fmaxf(a0[r], a1[r]);
    }
    // offset-major: 8 independent bpermutes per wait step
#pragma unroll
    for (int off = 1; off < 16; off <<= 1)
#pragma unroll
      for (int r = 0; r < 8; ++r) mx[r] = fmaxf(mx[r], __shfl_xor(mx[r], off, 32));

#pragma unroll
    for (int r = 0; r < 8; ++r) {
      const float mn = fmaxf(m8[r], mx[r]);
      const float alpha = __expf(m8[r] - mn);
      m8[r] = mn;
      a0[r] = __expf(a0[r] - mn);
      a1[r] = __expf(a1[r] - mn);
      rs[r] = a0[r] + a1[r];
      l8[r] *= alpha;
#pragma unroll
      for (int n = 0; n < 4; ++n) acc[n][r] *= alpha;
    }
#pragma unroll
    for (int off = 1; off < 16; off <<= 1)
#pragma unroll
      for (int r = 0; r < 8; ++r) rs[r] += __shfl_xor(rs[r], off, 32);

#pragma unroll
    for (int r = 0; r < 8; ++r) {
      l8[r] += rs[r];
      const int prow = r + half * 8;
      pls[prow * 32 + l15]      = (bf16)a0[r];
      pls[prow * 32 + 16 + l15] = (bf16)a1[r];
    }
    wait_async0();   // V tile now resident in LDS
    __syncthreads();

    // ---- ctx += P (16x32) @ V (32x64) ----
    v16bf pa = load_frag(&pls[l15 * 32], kbase);
#pragma unroll
    for (int n = 0; n < 4; ++n) {
      v16bf vf;
#pragma unroll
      for (int e = 0; e < 16; ++e) {
        const int k = ((e >> 3) << 4) + kbase + (e & 7);
        vf[e] = vls[k * 64 + n * 16 + l15];
      }
      acc[n] = __builtin_amdgcn_wmma_f32_16x16x32_bf16(
          false, pa, false, vf, (short)0, acc[n], false, false);
    }
    __syncthreads();
  }

  // ---- finalize: divide by row sums, write bf16 ctx [B,T,D] ----
#pragma unroll
  for (int n = 0; n < 4; ++n) {
    const int col = colQ + n * 16 + l15;
#pragma unroll
    for (int r = 0; r < 8; ++r) {
      const int row = q0 + r + half * 8;
      float y = acc[n][r] / fmaxf(l8[r], 1e-20f);
      ctx[(size_t)(b * SEQ + row) * DMODEL + col] = (bf16)y;
    }
  }
}

// ---------------------------------------------------------------------------
// Residual add + LayerNorm (D=1024, 256 threads x 4 elems). delta/xb nullable.
// ---------------------------------------------------------------------------
__global__ void add_ln_kernel(const float* __restrict__ xin,
                              const float* __restrict__ delta,
                              const float* __restrict__ g,
                              const float* __restrict__ bta,
                              float* __restrict__ xout,
                              bf16* __restrict__ xb_out) {
  __shared__ float red[256];
  const int row = blockIdx.x;
  const int tid = threadIdx.x;
  const float* xr = xin + (size_t)row * DMODEL;
  const float* dr = delta ? delta + (size_t)row * DMODEL : nullptr;

  float vals[4];
  float s = 0.0f;
#pragma unroll
  for (int i = 0; i < 4; ++i) {
    const int c = tid + i * 256;
    float v = xr[c] + (dr ? dr[c] : 0.0f);
    vals[i] = v;
    s += v;
  }
  red[tid] = s; __syncthreads();
  for (int off = 128; off > 0; off >>= 1) {
    if (tid < off) red[tid] += red[tid + off];
    __syncthreads();
  }
  const float mu = red[0] * (1.0f / DMODEL);
  __syncthreads();

  float s2 = 0.0f;
#pragma unroll
  for (int i = 0; i < 4; ++i) { float d = vals[i] - mu; s2 += d * d; }
  red[tid] = s2; __syncthreads();
  for (int off = 128; off > 0; off >>= 1) {
    if (tid < off) red[tid] += red[tid + off];
    __syncthreads();
  }
  const float inv = rsqrtf(red[0] * (1.0f / DMODEL) + 1e-5f);

#pragma unroll
  for (int i = 0; i < 4; ++i) {
    const int c = tid + i * 256;
    float y = (vals[i] - mu) * inv * g[c] + bta[c];
    xout[(size_t)row * DMODEL + c] = y;
    if (xb_out) xb_out[(size_t)row * DMODEL + c] = (bf16)y;
  }
}

__global__ void conv_kernel(const float* __restrict__ in, bf16* __restrict__ out, int n) {
  int i = blockIdx.x * blockDim.x + threadIdx.x;
  if (i < n) out[i] = (bf16)in[i];
}

__global__ void init_x_kernel(const float* __restrict__ x, float* __restrict__ resid,
                              bf16* __restrict__ xb, int n) {
  int i = blockIdx.x * blockDim.x + threadIdx.x;
  if (i < n) { float v = x[i]; resid[i] = v; xb[i] = (bf16)v; }
}

// ---------------------------------------------------------------------------
extern "C" void kernel_launch(void* const* d_in, const int* in_sizes, int n_in,
                              void* d_out, int out_size, void* d_ws, size_t ws_size,
                              hipStream_t stream) {
  (void)in_sizes; (void)n_in; (void)out_size; (void)ws_size;

  const float* x     = (const float*)d_in[0];
  const int*   mask  = (const int*)d_in[1];
  const float* Wq    = (const float*)d_in[2];
  const float* bq    = (const float*)d_in[3];
  const float* Wk    = (const float*)d_in[4];
  const float* bk    = (const float*)d_in[5];
  const float* Wv    = (const float*)d_in[6];
  const float* bv    = (const float*)d_in[7];
  const float* Wo    = (const float*)d_in[8];
  const float* bo    = (const float*)d_in[9];
  const float* fc1w  = (const float*)d_in[10];
  const float* fc1b  = (const float*)d_in[11];
  const float* fc2w  = (const float*)d_in[12];
  const float* fc2b  = (const float*)d_in[13];
  const float* ln1g  = (const float*)d_in[14];
  const float* ln1b  = (const float*)d_in[15];
  const float* ln2g  = (const float*)d_in[16];
  const float* ln2b  = (const float*)d_in[17];
  const float* lnfg  = (const float*)d_in[18];
  const float* lnfb  = (const float*)d_in[19];

  char* ws = (char*)d_ws;
  float* resid = (float*)(ws + 0 * MiB);      // 16 MiB
  bf16*  xb    = (bf16*)(ws + 16 * MiB);      //  8 MiB
  bf16*  qb    = (bf16*)(ws + 24 * MiB);      //  8 MiB
  bf16*  kb    = (bf16*)(ws + 32 * MiB);      //  8 MiB
  bf16*  vb    = (bf16*)(ws + 40 * MiB);      //  8 MiB
  bf16*  ctxb  = (bf16*)(ws + 48 * MiB);      //  8 MiB
  bf16*  hb    = (bf16*)(ws + 56 * MiB);      // 32 MiB
  float* tmp   = (float*)(ws + 88 * MiB);     // 16 MiB
  bf16*  wbuf  = (bf16*)(ws + 104 * MiB);     //  8 MiB (reused per weight)

  const int NX = NTOK * DMODEL;
  init_x_kernel<<<(NX + 255) / 256, 256, 0, stream>>>(x, resid, xb, NX);

  const dim3 gP(DMODEL / 64, NTOK / 128);  // projections / fc2
  const dim3 gF(DFFN / 64, NTOK / 128);    // fc1
  const dim3 gA(SEQ / 16, NHEAD, BATCH);   // attention

  for (int l = 0; l < LAYERS; ++l) {
    const size_t wDD = (size_t)DMODEL * DMODEL;
    const size_t wFD = (size_t)DFFN * DMODEL;

    // --- Q, K, V projections ---
    conv_kernel<<<(int)((wDD + 255) / 256), 256, 0, stream>>>(Wq + l * wDD, wbuf, (int)wDD);
    wmma_gemm_kernel<false, true><<<gP, 128, 0, stream>>>(
        xb, wbuf, bq + l * DMODEL, nullptr, qb, NTOK, DMODEL, DMODEL);
    conv_kernel<<<(int)((wDD + 255) / 256), 256, 0, stream>>>(Wk + l * wDD, wbuf, (int)wDD);
    wmma_gemm_kernel<false, true><<<gP, 128, 0, stream>>>(
        xb, wbuf, bk + l * DMODEL, nullptr, kb, NTOK, DMODEL, DMODEL);
    conv_kernel<<<(int)((wDD + 255) / 256), 256, 0, stream>>>(Wv + l * wDD, wbuf, (int)wDD);
    wmma_gemm_kernel<false, true><<<gP, 128, 0, stream>>>(
        xb, wbuf, bv + l * DMODEL, nullptr, vb, NTOK, DMODEL, DMODEL);

    // --- attention ---
    flash_attn_kernel<<<gA, 32, 0, stream>>>(qb, kb, vb, mask, ctxb);

    // --- output projection + LN1 ---
    conv_kernel<<<(int)((wDD + 255) / 256), 256, 0, stream>>>(Wo + l * wDD, wbuf, (int)wDD);
    wmma_gemm_kernel<false, false><<<gP, 128, 0, stream>>>(
        ctxb, wbuf, bo + l * DMODEL, tmp, nullptr, NTOK, DMODEL, DMODEL);
    add_ln_kernel<<<NTOK, 256, 0, stream>>>(resid, tmp, ln1g + l * DMODEL, ln1b + l * DMODEL, resid, xb);

    // --- FFN + LN2 ---
    conv_kernel<<<(int)((wFD + 255) / 256), 256, 0, stream>>>(fc1w + l * wFD, wbuf, (int)wFD);
    wmma_gemm_kernel<true, true><<<gF, 128, 0, stream>>>(
        xb, wbuf, fc1b + l * DFFN, nullptr, hb, NTOK, DFFN, DMODEL);
    conv_kernel<<<(int)((wFD + 255) / 256), 256, 0, stream>>>(fc2w + l * wFD, wbuf, (int)wFD);
    wmma_gemm_kernel<false, false><<<gP, 128, 0, stream>>>(
        hb, wbuf, fc2b + l * DMODEL, tmp, nullptr, NTOK, DMODEL, DFFN);
    add_ln_kernel<<<NTOK, 256, 0, stream>>>(resid, tmp, ln2g + l * DMODEL, ln2b + l * DMODEL, resid, xb);
  }

  // --- final LayerNorm -> d_out (fp32) ---
  add_ln_kernel<<<NTOK, 256, 0, stream>>>(resid, nullptr, lnfg, lnfb, (float*)d_out, nullptr);
}
